// DGCNN_9388798509199
// MI455X (gfx1250) — matrix operand order repaired and checked by reading, not compiled, not run on hardware
//
#include <hip/hip_runtime.h>
#include <hip/hip_bf16.h>

#define NUM_GRAPHS 1024
#define NPG 128                 // nodes per graph
#define N_NODES (NUM_GRAPHS * NPG)
#define EPG 2048                // edges per graph (16*128)
#define N_EDGES (N_NODES * 16)
#define HID 128
#define K_POOL 64
#define FEAT_C 385
#define C1K 416                 // conv1 K padded to 13*32

typedef __attribute__((ext_vector_type(16))) _Float16 v16h;
typedef __attribute__((ext_vector_type(8)))  float    v8f;
typedef __attribute__((ext_vector_type(4)))  unsigned u32x4;
typedef __attribute__((ext_vector_type(8)))  int      i32x8;
typedef __attribute__((ext_vector_type(4)))  int      i32x4;

#if defined(__HIP_DEVICE_COMPILE__) && __has_builtin(__builtin_amdgcn_tensor_load_to_lds) && __has_builtin(__builtin_amdgcn_s_wait_tensorcnt)
#define USE_TDM 1
#else
#define USE_TDM 0
#endif

union FragH { v16h v; unsigned u[8]; };
union FragF { v8f  v; float    f[8]; };

// ---- WMMA fragment helpers (CDNA5 16x16x32 f16 layouts, wave32) ------------

// A matrix 16x32 (MxK): lane m = L&15; K halves split on L>>4.
__device__ __forceinline__ v16h load_a_lds(const _Float16* S, int ld, int m0, int k0, int lane) {
    FragH r;
    const int m  = m0 + (lane & 15);
    const int kb = k0 + 8 * (lane >> 4);
    const _Float16* row = S + m * ld;
#pragma unroll
    for (int v = 0; v < 8; ++v) {
        const int k = kb + 16 * (v >> 2) + 2 * (v & 3);
        r.u[v] = *(const unsigned*)(row + k);   // two packed f16
    }
    return r.v;
}

// B matrix 32x16 (KxN): lane n = L&15; K halves split on L>>4 (same K order as A).
__device__ __forceinline__ v16h load_b_lds(const _Float16* S, int ld, int k0, int n0, int lane) {
    FragH r;
    const int n  = n0 + (lane & 15);
    const int kb = k0 + 8 * (lane >> 4);
#pragma unroll
    for (int s = 0; s < 16; ++s)
        r.v[s] = S[(kb + 16 * (s >> 3) + (s & 7)) * ld + n];
    return r.v;
}

// C/D 16x16 f32: VGPR v -> M = m0 + v + 8*(lane>>4), N = n0 + (lane&15).
__device__ __forceinline__ void store_d_lds_f16(_Float16* S, int ld, int m0, int n0, int lane, v8f c) {
    FragF u; u.v = c;
    const int n  = n0 + (lane & 15);
    const int mb = m0 + 8 * (lane >> 4);
#pragma unroll
    for (int v = 0; v < 8; ++v)
        S[(mb + v) * ld + n] = (_Float16)u.f[v];
}

// ---- Degree / normalization -------------------------------------------------

__global__ __launch_bounds__(256) void k_deg_init(float* deg) {
    int i = blockIdx.x * 256 + threadIdx.x;
    if (i < N_NODES) deg[i] = 1.0f;           // self loop
}

__global__ __launch_bounds__(256) void k_deg_edges(const int* __restrict__ ei, float* deg) {
    int e = blockIdx.x * 256 + threadIdx.x;
    if (e < N_EDGES) atomicAdd(&deg[ei[N_EDGES + e]], 1.0f);  // dst row
}

__global__ __launch_bounds__(256) void k_dinv(float* deg) {
    int i = blockIdx.x * 256 + threadIdx.x;
    if (i < N_NODES) deg[i] = rsqrtf(deg[i]);  // in place -> dinv
}

// ---- Dense normalized adjacency per graph (f32 LDS accumulate -> f16) ------

__global__ __launch_bounds__(256) void k_build_adj(const int* __restrict__ ei,
                                                   const float* __restrict__ dinv,
                                                   _Float16* __restrict__ A16) {
    __shared__ float sAcc[NPG * NPG];          // 64 KB
    const int g = blockIdx.x, tid = threadIdx.x;
    for (int i = tid; i < NPG * NPG; i += 256) sAcc[i] = 0.0f;
    __syncthreads();
    if (tid < NPG) {                           // self loop norm = dinv^2
        float d = dinv[g * NPG + tid];
        atomicAdd(&sAcc[tid * NPG + tid], d * d);
    }
    const int* src = ei;
    const int* dst = ei + N_EDGES;
    const int e0 = g * EPG;
    for (int e = tid; e < EPG; e += 256) {
        const int s = src[e0 + e], d = dst[e0 + e];
        const float w = dinv[s] * dinv[d];
        atomicAdd(&sAcc[(d - g * NPG) * NPG + (s - g * NPG)], w);   // ds_add_f32
    }
    __syncthreads();
    for (int i = tid; i < NPG * NPG; i += 256)
        A16[(size_t)g * (NPG * NPG) + i] = (_Float16)sAcc[i];
}

// ---- Weight prep: pad + f16 convert ----------------------------------------

__global__ __launch_bounds__(256) void k_wprep(const float* __restrict__ W0,
                                               const float* __restrict__ W1,
                                               const float* __restrict__ W2,
                                               const float* __restrict__ W3,
                                               const float* __restrict__ c1w,  // [16][1][385]
                                               const float* __restrict__ mw1,  // [896][128]
                                               _Float16* W0p, _Float16* W1p,
                                               _Float16* W2p, _Float16* W3p,
                                               _Float16* W1t, _Float16* MW1p) {
    const int tid = threadIdx.x;
    for (int i = tid; i < 32 * 128; i += 256) {       // [32][128], rows>=3 zero
        int k = i >> 7, n = i & 127;
        W0p[i] = (_Float16)((k < 3) ? W0[k * 128 + n] : 0.0f);
    }
    for (int i = tid; i < 128 * 128; i += 256) W1p[i] = (_Float16)W1[i];
    for (int i = tid; i < 128 * 128; i += 256) W2p[i] = (_Float16)W2[i];
    for (int i = tid; i < 128 * 16; i += 256) {       // [128][16], cols>=1 zero
        int k = i >> 4, n = i & 15;
        W3p[i] = (_Float16)((n == 0) ? W3[k] : 0.0f);
    }
    for (int i = tid; i < C1K * 16; i += 256) {       // conv1 W^T: [416][16]
        int k = i >> 4, c = i & 15;
        W1t[i] = (_Float16)((k < FEAT_C) ? c1w[c * FEAT_C + k] : 0.0f);
    }
    for (int i = tid; i < 896 * 128; i += 256) MW1p[i] = (_Float16)mw1[i];
}

// ---- Fully unrolled WMMA strip: D(16 x NT*16) += A(16x KT*32) @ B ----------

template <int KT, int NT, bool FIRST_COL_ONLY>
__device__ __forceinline__ void strip_mm_store(const _Float16* __restrict__ Amat, int lda,
                                               const _Float16* __restrict__ Bmat, int ldb,
                                               _Float16* __restrict__ sHout,        // may be null
                                               _Float16* __restrict__ featRow,      // feat + row base
                                               const float* __restrict__ bg,
                                               _Float16* __restrict__ sT, bool toT,
                                               int m0, int lane) {
#pragma unroll
    for (int nt = 0; nt < NT; ++nt) {
        v8f acc = {};
#pragma unroll
        for (int kt = 0; kt < KT; ++kt) {
            v16h a = load_a_lds(Amat, lda, m0, kt * 32, lane);
            v16h b = load_b_lds(Bmat, ldb, kt * 32, nt * 16, lane);
            acc = __builtin_amdgcn_wmma_f32_16x16x32_f16(false, a, false, b,
                                                         (short)0, acc, false, false);
        }
        if (toT) {
            store_d_lds_f16(sT, NT * 16, m0, nt * 16, lane, acc);
        } else {
            FragF u; u.v = acc;
            const int n = nt * 16 + (lane & 15);
            const float bv = bg[FIRST_COL_ONLY ? 0 : n];
#pragma unroll
            for (int v = 0; v < 8; ++v) {
                const int m = m0 + v + 8 * (lane >> 4);
                const float val = tanhf(u.f[v] + bv);
                if (FIRST_COL_ONLY) {
                    if (n == 0) featRow[(size_t)m * FEAT_C] = (_Float16)val;
                } else {
                    sHout[m * HID + n] = (_Float16)val;
                    featRow[(size_t)m * FEAT_C + n] = (_Float16)val;
                }
            }
        }
    }
}

// ---- Fused 4-layer GCN per graph: T = H@W ; O = A@T ; tanh(O+b) ------------

__global__ __launch_bounds__(256, 1) void k_gcn(const float* __restrict__ x,
                                                const _Float16* __restrict__ A16,
                                                const _Float16* __restrict__ W0p,
                                                const _Float16* __restrict__ W1p,
                                                const _Float16* __restrict__ W2p,
                                                const _Float16* __restrict__ W3p,
                                                const float* __restrict__ b0,
                                                const float* __restrict__ b1,
                                                const float* __restrict__ b2,
                                                const float* __restrict__ b3,
                                                _Float16* __restrict__ feat16) {
    __shared__ __align__(16) _Float16 sA[NPG * NPG];   // declared first: LDS offset 0 (TDM dest)
    __shared__ __align__(16) _Float16 sH[NPG * HID];
    __shared__ __align__(16) _Float16 sW[HID * HID];
    __shared__ __align__(16) _Float16 sT[NPG * HID];   // 128 KB total / 320 KB WGP

    const int g = blockIdx.x, tid = threadIdx.x;
    const int lane = tid & 31, wave = tid >> 5;        // wave32: 8 waves
    const int m0 = wave * 16;

    // ---- stage adjacency tile via Tensor Data Mover (async DMA -> LDS) -----
#if USE_TDM
    if (tid < 32) {                                    // one wave issues; EXEC ignored by TDM
        const unsigned long long ga =
            (unsigned long long)(uintptr_t)(A16 + (size_t)g * (NPG * NPG));
        u32x4 g0;                                      // D# group0
        g0[0] = 1u;                                    // count=1 valid descriptor
        g0[1] = 0u;                                    // lds_addr = 0 (sA at LDS base)
        g0[2] = (unsigned)(ga & 0xFFFFFFFFu);          // global_addr[31:0]
        g0[3] = (unsigned)((ga >> 32) & 0x01FFFFFFu) | (2u << 30);  // addr[56:32] | type=2
        i32x8 g1;                                      // D# group1
        g1[0] = (int)(1u << 16);                       // data_size=1 (2 bytes)
        g1[1] = (int)(16384u << 16);                   // tensor_dim0[15:0] = 16384
        g1[2] = (int)(1u << 16);                       // dim0 hi=0 | tensor_dim1 = 1
        g1[3] = (int)(16384u << 16);                   // dim1 hi=0 | tile_dim0 = 16384
        g1[4] = 1;                                     // tile_dim1 = 1, tile_dim2 = 0
        g1[5] = 16384;                                 // tensor_dim0_stride lo
        g1[6] = 0;                                     // stride0 hi | stride1 lo
        g1[7] = 0;
        i32x4 g2 = {0, 0, 0, 0}, g3 = {0, 0, 0, 0};
        i32x8 g4 = {0, 0, 0, 0, 0, 0, 0, 0};          // 6-arg (clang-23) form
        __builtin_amdgcn_tensor_load_to_lds(g0, g1, g2, g3, g4, 0);
        __builtin_amdgcn_s_wait_tensorcnt(0);
    }
#else
    for (int i = tid; i < NPG * NPG; i += 256) sA[i] = A16[(size_t)g * (NPG * NPG) + i];
#endif
    for (int i = tid; i < NPG * HID; i += 256) sH[i] = (_Float16)0.0f;
    __syncthreads();
    for (int i = tid; i < NPG * 3; i += 256) {
        int node = i / 3, c = i - node * 3;
        sH[node * HID + c] = (_Float16)x[(size_t)(g * NPG + node) * 3 + c];
    }
    _Float16* featRowL0 = feat16 + (size_t)g * NPG * FEAT_C;

    // -------- layer 0: K=32 (padded), N=128 --------
    for (int i = tid; i < 32 * 128; i += 256) sW[i] = W0p[i];
    __syncthreads();
    strip_mm_store<1, 8, false>(sH, HID, sW, 128, nullptr, nullptr, b0, sT, true, m0, lane);
    __syncthreads();
    strip_mm_store<4, 8, false>(sA, NPG, sT, 128, sH, featRowL0 + 0, b0, nullptr, false, m0, lane);
    __syncthreads();
    // -------- layer 1: K=128, N=128 --------
    for (int i = tid; i < 128 * 128; i += 256) sW[i] = W1p[i];
    __syncthreads();
    strip_mm_store<4, 8, false>(sH, HID, sW, 128, nullptr, nullptr, b1, sT, true, m0, lane);
    __syncthreads();
    strip_mm_store<4, 8, false>(sA, NPG, sT, 128, sH, featRowL0 + 128, b1, nullptr, false, m0, lane);
    __syncthreads();
    // -------- layer 2: K=128, N=128 --------
    for (int i = tid; i < 128 * 128; i += 256) sW[i] = W2p[i];
    __syncthreads();
    strip_mm_store<4, 8, false>(sH, HID, sW, 128, nullptr, nullptr, b2, sT, true, m0, lane);
    __syncthreads();
    strip_mm_store<4, 8, false>(sA, NPG, sT, 128, sH, featRowL0 + 256, b2, nullptr, false, m0, lane);
    __syncthreads();
    // -------- layer 3: K=128, N=16 (col 0 live) --------
    for (int i = tid; i < 128 * 16; i += 256) sW[i] = W3p[i];
    __syncthreads();
    strip_mm_store<4, 1, false>(sH, HID, sW, 16, nullptr, nullptr, b3, sT, true, m0, lane);
    __syncthreads();
    strip_mm_store<4, 1, true>(sA, NPG, sT, 16, nullptr, featRowL0 + 384, b3, nullptr, false, m0, lane);
}

// ---- SortPooling: bitonic argsort of 128 keys per graph, keep top 64 -------

__global__ __launch_bounds__(128) void k_sort(const _Float16* __restrict__ feat16,
                                              int* __restrict__ order) {
    __shared__ float key[NPG];
    __shared__ int   sidx[NPG];
    const int g = blockIdx.x, t = threadIdx.x;
    key[t]  = -(float)feat16[(size_t)(g * NPG + t) * FEAT_C + 384]; // ascending on -h4
    sidx[t] = t;
    __syncthreads();
    for (int k = 2; k <= NPG; k <<= 1) {
        for (int j = k >> 1; j > 0; j >>= 1) {
            const int ixj = t ^ j;
            if (ixj > t) {
                const float a = key[t], b = key[ixj];
                const bool up = ((t & k) == 0);
                if (up ? (a > b) : (a < b)) {
                    key[t] = b; key[ixj] = a;
                    int tmp = sidx[t]; sidx[t] = sidx[ixj]; sidx[ixj] = tmp;
                }
            }
            __syncthreads();
        }
    }
    if (t < K_POOL) order[g * K_POOL + t] = sidx[t];
}

// ---- Conv1 as WMMA GEMM: pooled[64 x 416] @ W1t[416 x 16], ReLU, MaxPool --

__global__ __launch_bounds__(256) void k_conv1(const _Float16* __restrict__ feat16,
                                               const int* __restrict__ order,
                                               const _Float16* __restrict__ W1t,  // [416][16]
                                               const float* __restrict__ bc1,
                                               float* __restrict__ z) {           // [G][16][32]
    __shared__ __align__(16) _Float16 sP[K_POOL * C1K];   // 52 KB
    __shared__ __align__(16) _Float16 sWc[C1K * 16];      // 13 KB
    __shared__ float sY[K_POOL * 16];                     // 4 KB
    const int g = blockIdx.x, tid = threadIdx.x;
    const int lane = tid & 31, wave = tid >> 5;
    for (int i = tid; i < C1K * 16; i += 256) sWc[i] = W1t[i];
    for (int i = tid; i < K_POOL * C1K; i += 256) {
        int kk = i / C1K, j = i - kk * C1K;
        int node = order[g * K_POOL + kk];
        sP[i] = (j < FEAT_C) ? feat16[(size_t)(g * NPG + node) * FEAT_C + j] : (_Float16)0.0f;
    }
    __syncthreads();
    if (wave < 4) {                                       // 4 waves: M tiles of 64 rows
        const int m0 = wave * 16;
        v8f acc = {};
#pragma unroll
        for (int kt = 0; kt < 13; ++kt) {
            v16h a = load_a_lds(sP, C1K, m0, kt * 32, lane);
            v16h b = load_b_lds(sWc, 16, kt * 32, 0, lane);
            acc = __builtin_amdgcn_wmma_f32_16x16x32_f16(false, a, false, b,
                                                         (short)0, acc, false, false);
        }
        FragF u; u.v = acc;
        const int n  = lane & 15;                          // channel
        const int mb = m0 + 8 * (lane >> 4);               // node slot
        const float bv = bc1[n];
#pragma unroll
        for (int v = 0; v < 8; ++v)
            sY[(mb + v) * 16 + n] = fmaxf(u.f[v] + bv, 0.0f);
    }
    __syncthreads();
    for (int p = tid; p < 16 * 32; p += 256) {            // MaxPool1d(2,2) over node dim
        int c = p >> 5, t2 = p & 31;
        z[(size_t)g * 512 + c * 32 + t2] =
            fmaxf(sY[(2 * t2) * 16 + c], sY[(2 * t2 + 1) * 16 + c]);
    }
}

// ---- Conv2 (16->32, k=5) + ReLU -> flat16 [G][896] -------------------------

__global__ __launch_bounds__(128) void k_conv2(const float* __restrict__ z,
                                               const float* __restrict__ w2,   // [32][16][5]
                                               const float* __restrict__ bc2,
                                               _Float16* __restrict__ flat16) {
    __shared__ float sz[16 * 32];
    __shared__ float sw2[32 * 16 * 5];
    const int g = blockIdx.x, tid = threadIdx.x;
    for (int i = tid; i < 512; i += 128) sz[i] = z[(size_t)g * 512 + i];
    for (int i = tid; i < 32 * 16 * 5; i += 128) sw2[i] = w2[i];
    __syncthreads();
    for (int p = tid; p < 32 * 28; p += 128) {
        int c = p / 28, t2 = p - c * 28;
        float acc = bc2[c];
        for (int i = 0; i < 16; ++i)
#pragma unroll
            for (int s = 0; s < 5; ++s)
                acc += sz[i * 32 + t2 + s] * sw2[(c * 16 + i) * 5 + s];
        flat16[(size_t)g * 896 + c * 28 + t2] = (_Float16)fmaxf(acc, 0.0f);
    }
}

// ---- MLP1 as batched WMMA GEMM [1024,896]@[896,128] + ReLU, then MLP2 ------

__global__ __launch_bounds__(256, 1) void k_mlp(const _Float16* __restrict__ flat16,
                                                const _Float16* __restrict__ MW1p, // [896][128]
                                                const float* __restrict__ mb1,
                                                const float* __restrict__ mw2,     // [128][5]
                                                const float* __restrict__ mb2,
                                                float* __restrict__ out) {
    __shared__ __align__(16) _Float16 sX[128 * 128];   // 32 KB rows chunk
    __shared__ __align__(16) _Float16 sW[128 * 128];   // 32 KB weight chunk
    __shared__ __align__(16) _Float16 sHn[128 * 128];  // 32 KB hidden
    const int gBase = blockIdx.x * 128, tid = threadIdx.x;
    const int lane = tid & 31, wave = tid >> 5;
    const int m0 = wave * 16;

    v8f acc[8] = {};
    for (int chunk = 0; chunk < 7; ++chunk) {          // K = 896 = 7 * 128
        for (int i = tid; i < 128 * 128; i += 256) {
            int r = i >> 7, k = i & 127;
            sX[i] = flat16[(size_t)(gBase + r) * 896 + chunk * 128 + k];
        }
        for (int i = tid; i < 128 * 128; i += 256) {
            int k = i >> 7, n = i & 127;
            sW[i] = MW1p[(size_t)(chunk * 128 + k) * 128 + n];
        }
        __syncthreads();
#pragma unroll
        for (int nt = 0; nt < 8; ++nt) {
#pragma unroll
            for (int kt = 0; kt < 4; ++kt) {
                v16h a = load_a_lds(sX, 128, m0, kt * 32, lane);
                v16h b = load_b_lds(sW, 128, kt * 32, nt * 16, lane);
                acc[nt] = __builtin_amdgcn_wmma_f32_16x16x32_f16(false, a, false, b,
                                                                 (short)0, acc[nt], false, false);
            }
        }
        __syncthreads();
    }
    // ReLU epilogue -> hidden in LDS (f16)
#pragma unroll
    for (int nt = 0; nt < 8; ++nt) {
        FragF u; u.v = acc[nt];
        const int n = nt * 16 + (lane & 15);
        const float bv = mb1[n];
#pragma unroll
        for (int v = 0; v < 8; ++v) {
            const int m = m0 + v + 8 * (lane >> 4);
            sHn[m * 128 + n] = (_Float16)fmaxf(u.f[v] + bv, 0.0f);
        }
    }
    __syncthreads();
    // MLP2: 128 -> 5 per row
    for (int p = tid; p < 128 * 5; p += 256) {
        int row = p / 5, c = p - row * 5;
        float a2 = mb2[c];
        for (int o = 0; o < 128; ++o)
            a2 += (float)sHn[row * 128 + o] * mw2[o * 5 + c];
        out[(size_t)(gBase + row) * 5 + c] = a2;
    }
}

// ---- Host launcher ---------------------------------------------------------

static inline size_t alignUp(size_t v, size_t a) { return (v + a - 1) & ~(a - 1); }

extern "C" void kernel_launch(void* const* d_in, const int* in_sizes, int n_in,
                              void* d_out, int out_size, void* d_ws, size_t ws_size,
                              hipStream_t stream) {
    const float* x   = (const float*)d_in[0];
    const int*   ei  = (const int*)d_in[1];
    // d_in[2] = batch (implicit via layout; unused)
    const float* W0  = (const float*)d_in[3];
    const float* b0  = (const float*)d_in[4];
    const float* W1  = (const float*)d_in[5];
    const float* b1  = (const float*)d_in[6];
    const float* W2  = (const float*)d_in[7];
    const float* b2  = (const float*)d_in[8];
    const float* W3  = (const float*)d_in[9];
    const float* b3  = (const float*)d_in[10];
    const float* c1w = (const float*)d_in[11];
    const float* c1b = (const float*)d_in[12];
    const float* c2w = (const float*)d_in[13];
    const float* c2b = (const float*)d_in[14];
    const float* mw1 = (const float*)d_in[15];
    const float* mb1 = (const float*)d_in[16];
    const float* mw2 = (const float*)d_in[17];
    const float* mb2 = (const float*)d_in[18];
    float* out = (float*)d_out;

    // workspace carve
    char* p = (char*)d_ws;
    size_t off = 0;
    auto carve = [&](size_t bytes) { void* r = p + off; off = alignUp(off + bytes, 256); return r; };
    float*     deg    = (float*)carve((size_t)N_NODES * 4);                 // -> dinv in place
    _Float16*  A16    = (_Float16*)carve((size_t)NUM_GRAPHS * NPG * NPG * 2);
    _Float16*  W0p    = (_Float16*)carve((size_t)32 * 128 * 2);
    _Float16*  W1p    = (_Float16*)carve((size_t)128 * 128 * 2);
    _Float16*  W2p    = (_Float16*)carve((size_t)128 * 128 * 2);
    _Float16*  W3p    = (_Float16*)carve((size_t)128 * 16 * 2);
    _Float16*  W1t    = (_Float16*)carve((size_t)C1K * 16 * 2);
    _Float16*  MW1p   = (_Float16*)carve((size_t)896 * 128 * 2);
    _Float16*  feat16 = (_Float16*)carve((size_t)N_NODES * FEAT_C * 2);
    int*       order  = (int*)carve((size_t)NUM_GRAPHS * K_POOL * 4);
    float*     zbuf   = (float*)carve((size_t)NUM_GRAPHS * 512 * 4);
    _Float16*  flat16 = (_Float16*)carve((size_t)NUM_GRAPHS * 896 * 2);
    (void)ws_size; (void)n_in; (void)in_sizes; (void)out_size;

    k_deg_init <<<N_NODES / 256, 256, 0, stream>>>(deg);
    k_deg_edges<<<N_EDGES / 256, 256, 0, stream>>>(ei, deg);
    k_dinv     <<<N_NODES / 256, 256, 0, stream>>>(deg);
    k_build_adj<<<NUM_GRAPHS, 256, 0, stream>>>(ei, deg, A16);
    k_wprep    <<<1, 256, 0, stream>>>(W0, W1, W2, W3, c1w, mw1,
                                       W0p, W1p, W2p, W3p, W1t, MW1p);
    k_gcn      <<<NUM_GRAPHS, 256, 0, stream>>>(x, A16, W0p, W1p, W2p, W3p,
                                                b0, b1, b2, b3, feat16);
    k_sort     <<<NUM_GRAPHS, 128, 0, stream>>>(feat16, order);
    k_conv1    <<<NUM_GRAPHS, 256, 0, stream>>>(feat16, order, W1t, c1b, zbuf);
    k_conv2    <<<NUM_GRAPHS, 128, 0, stream>>>(zbuf, c2w, c2b, flat16);
    k_mlp      <<<NUM_GRAPHS / 128, 256, 0, stream>>>(flat16, MW1p, mb1, mw2, mb2, out);
}
